// GraphConvolution_15290083574330
// MI455X (gfx1250) — matrix operand order repaired and checked by reading, not compiled, not run on hardware
//
#include <hip/hip_runtime.h>
#include <math.h>

typedef __attribute__((ext_vector_type(16))) _Float16 v16h;
typedef __attribute__((ext_vector_type(8)))  float    v8f;

#define FDIM 64

// ---------------------------------------------------------------------------
// Zero the scatter accumulators (must be 0 before atomics every call)
// ---------------------------------------------------------------------------
__global__ __launch_bounds__(256) void zero_kernel(float* __restrict__ p, long long total) {
  long long i = (long long)blockIdx.x * blockDim.x + threadIdx.x;
  if (i < total) p[i] = 0.f;
}

// ---------------------------------------------------------------------------
// Fused triple GEMM: Hl = x@Wl, Hh = x@Wh, Hm = relu(x@Wm)
// Block = 256 threads = 8 waves. Block tile = 128 rows; wave tile = 16 rows x 64 cols.
// WMMA f32_16x16x32_f16, K=64 split into two K=32 chunks.
// Weights stored TRANSPOSED in LDS so every fragment half-pair (k,k+1) is a
// contiguous aligned u32 -> compiler merges gathers into ds_load_b128.
// ---------------------------------------------------------------------------
__global__ __launch_bounds__(256) void gemm3_wmma_kernel(
    const float* __restrict__ x,
    const float* __restrict__ Wl, const float* __restrict__ Wh, const float* __restrict__ Wm,
    float* __restrict__ Hl, float* __restrict__ Hh, float* __restrict__ Hm, int n)
{
  __shared__ _Float16 sWT[3][FDIM][FDIM + 8];  // sWT[w][n][k] = W[k][n] (transposed)
  __shared__ _Float16 sX [128][FDIM + 8];      // x row tile

  const int tid = threadIdx.x;

  // Stage all three 64x64 weight matrices into LDS as f16, transposed
  for (int i = tid; i < 3 * FDIM * FDIM; i += 256) {
    int w = i / (FDIM * FDIM);
    int r = (i / FDIM) % FDIM;   // k index in W
    int c = i % FDIM;            // n index in W
    const float* W = (w == 0) ? Wl : (w == 1 ? Wh : Wm);
    sWT[w][c][r] = (_Float16)W[r * FDIM + c];
  }

  // Stage 128 rows of x into LDS as f16 (zero-pad past n)
  const long long rowbase = (long long)blockIdx.x * 128;
  for (int i = tid; i < 128 * FDIM; i += 256) {
    int r = i / FDIM, c = i % FDIM;
    long long row = rowbase + r;
    sX[r][c] = (row < n) ? (_Float16)x[row * FDIM + c] : (_Float16)0.f;
  }
  __syncthreads();

  const int wave = tid >> 5;
  const int lane = tid & 31;
  const int m    = lane & 15;    // A: row within tile / B,C: column within N-tile
  const int hi   = lane >> 4;
  const int wrow = wave * 16;    // first local row of this wave's 16-row tile

  union HF { v16h v; unsigned int u[8]; };

  // A fragments: 16x32 f16 layout (ISA 7.12.2). u[v] packs halves (kk, kk+1):
  // v<4: kk = hi*8 + 2v ; v>=4: kk = 16 + hi*8 + 2(v-4). All kk even -> aligned u32.
  HF a[2];
  {
    const unsigned int* xrow = (const unsigned int*)&sX[wrow + m][0];  // u32 i = halves 2i,2i+1
    #pragma unroll
    for (int kc = 0; kc < 2; ++kc) {
      #pragma unroll
      for (int v = 0; v < 8; ++v) {
        int kk = (v < 4) ? (hi * 8 + 2 * v) : (16 + hi * 8 + 2 * (v - 4));
        a[kc].u[v] = xrow[(kc * 32 + kk) >> 1];
      }
    }
  }

  const bool fulltile = (rowbase + 128 <= (long long)n);

  #pragma unroll
  for (int w = 0; w < 3; ++w) {
    float* H = (w == 0) ? Hl : (w == 1 ? Hh : Hm);
    #pragma unroll
    for (int jt = 0; jt < 4; ++jt) {          // four 16-wide N tiles
      const unsigned int* wcol = (const unsigned int*)&sWT[w][jt * 16 + m][0];
      v8f c = {};
      #pragma unroll
      for (int kc = 0; kc < 2; ++kc) {        // two 32-deep K chunks
        // B fragment: 32x16 f16; u32 r holds kk, kk+1 with kk = hi*8 + 2r (r<4), 16 + hi*8 + 2(r-4) (r>=4).
        // In sWT a column of W is a contiguous row -> u32 loads, mergeable to b128.
        HF b;
        #pragma unroll
        for (int r = 0; r < 8; ++r)
          b.u[r] = wcol[(kc * 32 + ((r < 4) ? (hi * 8 + 2 * r) : (16 + hi * 8 + 2 * (r - 4)))) >> 1];
        c = __builtin_amdgcn_wmma_f32_16x16x32_f16(
                /*neg_a=*/false, a[kc].v, /*neg_b=*/false, b.v,
                /*c_mod=*/(short)0, c, /*reuse_a=*/false, /*reuse_b=*/false);
      }
      // C/D layout: VGPR r -> row (hi*8 + r), col = jt*16 + m
      float* Hp = H + (rowbase + wrow + hi * 8) * FDIM + jt * 16 + m;
      if (fulltile) {
        #pragma unroll
        for (int r = 0; r < 8; ++r) {
          float vv = c[r];
          if (w == 2) vv = fmaxf(vv, 0.f);     // ReLU fused for MLP branch
          Hp[r * FDIM] = vv;
        }
      } else {
        long long row0 = rowbase + wrow + hi * 8;
        #pragma unroll
        for (int r = 0; r < 8; ++r) {
          if (row0 + r < n) {
            float vv = c[r];
            if (w == 2) vv = fmaxf(vv, 0.f);
            Hp[r * FDIM] = vv;
          }
        }
      }
    }
  }
}

// ---------------------------------------------------------------------------
// COO SpMM scatter: agg[row] += val * H[col]  (one thread per edge x float4)
// ---------------------------------------------------------------------------
__global__ __launch_bounds__(256) void spmm_scatter_kernel(
    const int* __restrict__ rowi, const int* __restrict__ coli, const float* __restrict__ val,
    const float* __restrict__ H, float* __restrict__ agg, int E)
{
  long long gid = (long long)blockIdx.x * blockDim.x + threadIdx.x;
  long long total = (long long)E * 16;
  if (gid >= total) return;
  int e = (int)(gid >> 4);
  int q = (int)(gid & 15);                 // which float4 of the 64-feature row
  int r = rowi[e];
  int c = coli[e];
  float w = val[e];
  const float4 hv = ((const float4*)H)[(long long)c * 16 + q];
  float* dst = agg + (long long)r * FDIM + q * 4;
  atomicAdd(dst + 0, w * hv.x);
  atomicAdd(dst + 1, w * hv.y);
  atomicAdd(dst + 2, w * hv.z);
  atomicAdd(dst + 3, w * hv.w);
}

// ---------------------------------------------------------------------------
// Per-node attention mix: one wave (32 lanes) per node, lane owns features
// {lane, lane+32}. Three dot-products reduced via wave32 xor-butterfly.
// ---------------------------------------------------------------------------
__global__ __launch_bounds__(256) void finalize_kernel(
    const float* __restrict__ aggL, const float* __restrict__ aggH, const float* __restrict__ Hm,
    const float* __restrict__ aL, const float* __restrict__ aH, const float* __restrict__ aM,
    const float* __restrict__ att3, float* __restrict__ out, int n)
{
  long long gtid = (long long)blockIdx.x * blockDim.x + threadIdx.x;
  int node = (int)(gtid >> 5);
  int lane = (int)(gtid & 31);
  if (node >= n) return;
  long long base = (long long)node * FDIM;

  float l0 = fmaxf(aggL[base + lane],      0.f);   // out_low = relu(agg_low)
  float l1 = fmaxf(aggL[base + lane + 32], 0.f);
  float h0 = fmaxf(aggH[base + lane],      0.f);   // out_high
  float h1 = fmaxf(aggH[base + lane + 32], 0.f);
  float m0 = Hm[base + lane];                       // out_mlp (already relu'd)
  float m1 = Hm[base + lane + 32];

  float dl = l0 * aL[lane] + l1 * aL[lane + 32];
  float dh = h0 * aH[lane] + h1 * aH[lane + 32];
  float dm = m0 * aM[lane] + m1 * aM[lane + 32];
  #pragma unroll
  for (int off = 16; off > 0; off >>= 1) {          // wave32 butterfly reduce
    dl += __shfl_xor(dl, off, 32);
    dh += __shfl_xor(dh, off, 32);
    dm += __shfl_xor(dm, off, 32);
  }

  float s0 = 1.f / (1.f + expf(-dl));
  float s1 = 1.f / (1.f + expf(-dh));
  float s2 = 1.f / (1.f + expf(-dm));

  float lg[3];
  #pragma unroll
  for (int j = 0; j < 3; ++j)
    lg[j] = (s0 * att3[0 * 3 + j] + s1 * att3[1 * 3 + j] + s2 * att3[2 * 3 + j]) * (1.f / 3.f);

  float mx = fmaxf(lg[0], fmaxf(lg[1], lg[2]));
  float e0 = expf(lg[0] - mx), e1 = expf(lg[1] - mx), e2 = expf(lg[2] - mx);
  float inv = 1.f / (e0 + e1 + e2);
  float a0 = e0 * inv, a1 = e1 * inv, a2 = e2 * inv;

  out[base + lane]      = 3.f * (a0 * l0 + a1 * h0 + a2 * m0);
  out[base + lane + 32] = 3.f * (a0 * l1 + a1 * h1 + a2 * m1);
}

// ---------------------------------------------------------------------------
extern "C" void kernel_launch(void* const* d_in, const int* in_sizes, int n_in,
                              void* d_out, int out_size, void* d_ws, size_t ws_size,
                              hipStream_t stream)
{
  const float* x        = (const float*)d_in[0];
  const int*   row_low  = (const int*)  d_in[1];
  const int*   col_low  = (const int*)  d_in[2];
  const float* val_low  = (const float*)d_in[3];
  const int*   row_high = (const int*)  d_in[4];
  const int*   col_high = (const int*)  d_in[5];
  const float* val_high = (const float*)d_in[6];
  const float* W_low    = (const float*)d_in[7];
  const float* W_high   = (const float*)d_in[8];
  const float* W_mlp    = (const float*)d_in[9];
  const float* a_low    = (const float*)d_in[10];
  const float* a_high   = (const float*)d_in[11];
  const float* a_mlp    = (const float*)d_in[12];
  const float* att3     = (const float*)d_in[13];
  float* out = (float*)d_out;

  const int n = in_sizes[0] / FDIM;   // 50000
  const int E = in_sizes[1];          // 800000

  // Workspace layout: Hl | Hh | Hm | aggL | aggH  (5 * n * 64 floats = 64 MB)
  float* ws = (float*)d_ws;
  long long nf = (long long)n * FDIM;
  float* Hl   = ws;
  float* Hh   = ws + nf;
  float* Hm   = ws + 2 * nf;
  float* aggL = ws + 3 * nf;
  float* aggH = ws + 4 * nf;

  {  // aggL & aggH are contiguous: zero both in one launch
    long long total = 2 * nf;
    int blocks = (int)((total + 255) / 256);
    zero_kernel<<<blocks, 256, 0, stream>>>(aggL, total);
  }
  {
    int blocks = (n + 127) / 128;
    gemm3_wmma_kernel<<<blocks, 256, 0, stream>>>(x, W_low, W_high, W_mlp, Hl, Hh, Hm, n);
  }
  {
    long long total = (long long)E * 16;
    int blocks = (int)((total + 255) / 256);
    spmm_scatter_kernel<<<blocks, 256, 0, stream>>>(row_low,  col_low,  val_low,  Hl, aggL, E);
    spmm_scatter_kernel<<<blocks, 256, 0, stream>>>(row_high, col_high, val_high, Hh, aggH, E);
  }
  {
    long long total = (long long)n * 32;
    int blocks = (int)((total + 255) / 256);
    finalize_kernel<<<blocks, 256, 0, stream>>>(aggL, aggH, Hm, a_low, a_high, a_mlp, att3, out, n);
  }
}
